// MultiSourceAttentionBlock_41223096107758
// MI455X (gfx1250) — matrix operand order repaired and checked by reading, not compiled, not run on hardware
//
#include <hip/hip_runtime.h>
#include <math.h>

// ---------------------------------------------------------------------------
// MultiSourceAttentionBlock forward for MI455X (gfx1250, wave32, WMMA).
// Memory-bound layer (~220MB HBM traffic vs ~180 GFLOP): fp32 in HBM,
// f16-in/f32-accum WMMA (v_wmma_f32_16x16x32_f16) for all GEMMs.
// 64x128 block tile, wave = 16x64 (4 WMMAs / A-fragment), double-buffered
// b128 staging + global_prefetch_b8.
// ---------------------------------------------------------------------------

typedef _Float16 half_t;
typedef _Float16 v16h __attribute__((ext_vector_type(16)));
typedef _Float16 v8h  __attribute__((ext_vector_type(8)));
typedef float    v8f  __attribute__((ext_vector_type(8)));
typedef float    v4f  __attribute__((ext_vector_type(4)));

#define D_   1024
#define H_   16
#define HD_  64
#define FF_  4096
#define B_   16
#define T_   256
#define TBEV_ 64
#define TR_  512

static __device__ __forceinline__ void store_h8(half_t* dst, v4f x, v4f y) {
  v8h h;
  #pragma unroll
  for (int j = 0; j < 4; ++j) { h[j] = (half_t)x[j]; h[4 + j] = (half_t)y[j]; }
  *(v8h*)dst = h;  // 16B aligned -> single ds_store_b128
}

// ======================= generic WMMA GEMM =================================
// C[M,N] = act(A[M,K] @ Wseg[N,K]^T + bseg[N]) (+src) (*tanh(gate)) ...
// Block: 256 threads = 8 waves; 64(M) x 128(N) tile; K staged 32-wide as f16.
// Wave w: rows 16*(w>>1), cols 64*(w&1); 4 WMMAs per chunk reusing one A frag.
__global__ __launch_bounds__(256) void gemm_wmma(
    const float* __restrict__ A, int M, int N, int K,
    const float* __restrict__ W0, const float* __restrict__ W1,
    const float* __restrict__ b0, const float* __restrict__ b1,
    const int*  __restrict__ Ttraj,
    float* __restrict__ out,
    const float* __restrict__ src,          // optional add, row-major MxN
    int act,                                 // 0 none, 1 tanh, 2 gelu(exact)
    const float* __restrict__ gate_scalar,   // optional *tanh(*p)
    const float* __restrict__ res,           // optional: out = res + g_mod*val
    const float* __restrict__ modv, const float* __restrict__ rmodv, int slice,
    int head_L)                              // >0: store (B,H,L,HD) layout
{
  __shared__ __align__(16) half_t As[64][40];
  __shared__ __align__(16) half_t Ws[128][40];

  const int m0 = blockIdx.y * 64;
  const int n0 = blockIdx.x * 128;
  const int tt = Ttraj ? *Ttraj : 192;

  // 64-row tiles never straddle the traj/rest boundary (64 | 192, 64 | 256).
  const bool segTraj = ((m0 % T_) < tt);
  const float* W    = (W1 && !segTraj) ? W1 : W0;
  const float* bias = (b1 && !segTraj) ? b1 : b0;

  const int tid  = threadIdx.x;
  const int lane = tid & 31;
  const int wave = tid >> 5;
  const int wm = wave >> 1, wn2 = wave & 1;
  const int hi = lane >> 4, ln = lane & 15;

  v8f acc[4];
  #pragma unroll
  for (int t2 = 0; t2 < 4; ++t2)
    #pragma unroll
    for (int i = 0; i < 8; ++i) acc[t2][i] = 0.f;

  // ---- staging geometry ----------------------------------------------------
  // A: thread stages 8 consecutive floats of one of 64 rows.
  const int lr = tid >> 2;
  const int lk = (tid & 3) * 8;
  const int gr = m0 + lr;
  const int grc = (gr < M) ? gr : (M - 1);
  const float sel = (gr < M) ? 1.f : 0.f;
  const bool fullM = (m0 + 64 <= M);          // block-uniform: skip sel mul
  const float* Arow = A + (size_t)grc * K + lk;
  // W: thread stages 16 consecutive floats of one of 128 rows (N%128==0).
  const int lrw = tid >> 1;
  const int lkw = (tid & 1) * 16;
  const float* Wrow = W + (size_t)(n0 + lrw) * K + lkw;

  // prologue fetch (chunk 0)
  v4f a0 = ((const v4f*)Arow)[0];
  v4f a1 = ((const v4f*)Arow)[1];
  if (!fullM) { a0 *= sel; a1 *= sel; }
  v4f w0 = ((const v4f*)Wrow)[0];
  v4f w1 = ((const v4f*)Wrow)[1];
  v4f w2 = ((const v4f*)Wrow)[2];
  v4f w3 = ((const v4f*)Wrow)[3];

  for (int k0 = 0; k0 < K; k0 += 32) {
    store_h8(&As[lr][lk], a0, a1);
    store_h8(&Ws[lrw][lkw], w0, w1);
    store_h8(&Ws[lrw][lkw + 8], w2, w3);
    __syncthreads();

    if (k0 + 32 < K) {  // software pipeline: fetch next chunk during WMMA
      const float* an = Arow + k0 + 32;
      const float* wn = Wrow + k0 + 32;
      a0 = ((const v4f*)an)[0];
      a1 = ((const v4f*)an)[1];
      if (!fullM) { a0 *= sel; a1 *= sel; }
      w0 = ((const v4f*)wn)[0];
      w1 = ((const v4f*)wn)[1];
      w2 = ((const v4f*)wn)[2];
      w3 = ((const v4f*)wn)[3];
      if (k0 + 64 < K) {  // prefetch chunk after next toward L2
        __builtin_prefetch(Arow + k0 + 64, 0, 0);
        __builtin_prefetch(Wrow + k0 + 64, 0, 0);
      }
    }

    // A fragment: lane holds row m, K pairs per ISA 7.12.2 layout
    v16h af;
    const int m = wm * 16 + ln;
    #pragma unroll
    for (int v = 0; v < 8; ++v) {
      const int k = ((v >> 2) << 4) + (hi << 3) + ((v & 3) << 1);
      af[2 * v]     = As[m][k];
      af[2 * v + 1] = As[m][k + 1];
    }
    // 4 B fragments (cols wn2*64 + t2*16), one WMMA each, A reused.
    #pragma unroll
    for (int t2 = 0; t2 < 4; ++t2) {
      v16h bf;
      const int n = wn2 * 64 + t2 * 16 + ln;
      #pragma unroll
      for (int e = 0; e < 16; ++e) bf[e] = Ws[n][((e >> 3) << 4) + (hi << 3) + (e & 7)];
      acc[t2] = __builtin_amdgcn_wmma_f32_16x16x32_f16(
          false, af, false, bf, (short)0, acc[t2], false, false);
    }
    __syncthreads();
  }

  // --------------------------- epilogue ------------------------------------
  const float gs = gate_scalar ? tanhf(*gate_scalar) : 1.f;
  #pragma unroll
  for (int t2 = 0; t2 < 4; ++t2) {
    const int col = n0 + wn2 * 64 + t2 * 16 + ln;
    #pragma unroll
    for (int v = 0; v < 8; ++v) {
      const int row = m0 + wm * 16 + v + 8 * hi;
      if (row >= M) continue;
      float val = acc[t2][v] + (bias ? bias[col] : 0.f);
      if (act == 1)      val = tanhf(val);
      else if (act == 2) val = 0.5f * val * (1.f + erff(val * 0.70710678f));
      if (src) val += src[(size_t)row * N + col];
      val *= gs;
      if (res) {  // adaLN-gated residual (N == D here)
        const int b = row / T_, t = row % T_;
        const float* mm = (t < tt) ? modv : rmodv;
        const float gmod = mm[(size_t)b * (6 * D_) + (size_t)slice * D_ + col];
        val = res[(size_t)row * N + col] + gmod * val;
      }
      if (head_L > 0) { // (B,H,L,HD) store; N == D
        const int b = row / head_L, t = row % head_L;
        const int h = col >> 6, d = col & 63;
        out[(((size_t)(b * H_ + h)) * head_L + t) * HD_ + d] = val;
      } else {
        out[(size_t)row * N + col] = val;
      }
    }
  }
}

// ================== LayerNorm + adaLN shift/scale ==========================
__global__ __launch_bounds__(256) void ln_mod_kernel(
    const float* __restrict__ x, const float* __restrict__ w,
    const float* __restrict__ bln,
    const float* __restrict__ modv, const float* __restrict__ rmodv,
    int slice_scale, int slice_shift, const int* __restrict__ Ttraj,
    float* __restrict__ out)
{
  __shared__ float red[256];
  const int r = blockIdx.x, tid = threadIdx.x;
  const float* xr = x + (size_t)r * D_;
  float vals[4], s = 0.f;
  #pragma unroll
  for (int i = 0; i < 4; ++i) { vals[i] = xr[tid + i * 256]; s += vals[i]; }
  red[tid] = s; __syncthreads();
  for (int st = 128; st > 0; st >>= 1) {
    if (tid < st) red[tid] += red[tid + st];
    __syncthreads();
  }
  const float mean = red[0] * (1.f / D_);
  __syncthreads();
  float s2 = 0.f;
  #pragma unroll
  for (int i = 0; i < 4; ++i) { float d = vals[i] - mean; s2 += d * d; }
  red[tid] = s2; __syncthreads();
  for (int st = 128; st > 0; st >>= 1) {
    if (tid < st) red[tid] += red[tid + st];
    __syncthreads();
  }
  const float inv = rsqrtf(red[0] * (1.f / D_) + 1e-5f);
  const int b = r >> 8, t = r & 255;
  const float* mm = (t < *Ttraj) ? modv : rmodv;
  #pragma unroll
  for (int i = 0; i < 4; ++i) {
    const int c = tid + i * 256;
    const float xn  = (vals[i] - mean) * inv * w[c] + bln[c];
    const float scl = mm[(size_t)b * (6 * D_) + (size_t)slice_scale * D_ + c];
    const float shf = mm[(size_t)b * (6 * D_) + (size_t)slice_shift * D_ + c];
    out[(size_t)r * D_ + c] = xn * (1.f + scl) + shf;
  }
}

// ================== per-head RMSNorm (+ optional RoPE) =====================
// block = 64 threads = one head-row; grid = B*H*L.  In-place safe.
__global__ __launch_bounds__(64) void rms_rope_kernel(
    const float* __restrict__ src, float* __restrict__ dst, int L,
    const float* __restrict__ wvec, int do_rope, int src_rowmajor)
{
  __shared__ float row[64];
  __shared__ float red[64];
  const int idx = blockIdx.x;
  const int t = idx % L;
  const int bh = idx / L;
  const int h = bh % H_, b = bh / H_;
  const int d = threadIdx.x;
  float v;
  if (src_rowmajor) v = src[((size_t)(b * L + t)) * D_ + h * HD_ + d];
  else              v = src[(size_t)idx * HD_ + d];
  red[d] = v * v; __syncthreads();
  for (int st = 32; st > 0; st >>= 1) {
    if (d < st) red[d] += red[d + st];
    __syncthreads();
  }
  const float inv = rsqrtf(red[0] * (1.f / HD_) + 1e-6f);
  const float xr = v * inv * wvec[d];
  row[d] = xr; __syncthreads();
  float o = xr;
  if (do_rope) {
    const int j = d & 31;
    const float f = (float)t * __expf(-(float)j * 0.2878231366f); // ln(1e4)/32
    const float c = __cosf(f), sn = __sinf(f);
    o = (d < 32) ? (xr * c - row[d + 32] * sn) : (xr * c + row[d - 32] * sn);
  }
  dst[(size_t)idx * HD_ + d] = o;
}

// ================== flash-style attention ==================================
// grid = B*H; block = 256 (= Lq rows, one per thread); K/V chunks of 64 in LDS.
// All Ks/Vs compute reads are wave-broadcast -> no pad needed; fully float4.
__global__ __launch_bounds__(256) void attn_kernel(
    const float* __restrict__ Q, const float* __restrict__ Kv,
    const float* __restrict__ Vv, int Lk,
    const float* __restrict__ t0p, const float* __restrict__ b0p,
    const float* __restrict__ t1p, const float* __restrict__ b1p,
    const int*  __restrict__ Ttraj,
    const float* __restrict__ gscale_p, int mode /*0 write, 1 add*/,
    float* __restrict__ Out)
{
  extern __shared__ float sm[];
  float* Ks  = sm;                 // 64*64
  float* Vs  = sm + 64 * 64;       // 64*64
  float* Ssc = sm + 2 * 64 * 64;   // 256*64 score stash
  const int bh = blockIdx.x;
  const int h = bh % H_, b = bh / H_;
  const int t = threadIdx.x;

  const int tt = *Ttraj;
  float temp = 1.f + log1pf(expf(*t0p));
  float bi = *b0p;
  if (t1p && t >= tt) { temp = 1.f + log1pf(expf(*t1p)); bi = *b1p; }
  const float scale = 0.125f / temp;  // 1/sqrt(64) * inv_temp

  v4f q4[16];
  const v4f* qp = (const v4f*)(Q + (((size_t)bh) * T_ + t) * HD_);
  #pragma unroll
  for (int i = 0; i < 16; ++i) q4[i] = qp[i];

  float mrun = -1e30f, l = 0.f;
  v4f acc4[16];
  #pragma unroll
  for (int i = 0; i < 16; ++i) {
    #pragma unroll
    for (int j = 0; j < 4; ++j) acc4[i][j] = 0.f;
  }

  for (int kc = 0; kc < Lk; kc += 64) {
    // stage 64x64 K and V chunks (contiguous 16KB each) as float4
    const v4f* gk = (const v4f*)(Kv + (((size_t)bh) * Lk + kc) * HD_);
    const v4f* gv = (const v4f*)(Vv + (((size_t)bh) * Lk + kc) * HD_);
    for (int i = threadIdx.x; i < 1024; i += 256) {
      ((v4f*)Ks)[i] = gk[i];
      ((v4f*)Vs)[i] = gv[i];
    }
    __syncthreads();
    float cmax = -1e30f;
    for (int j = 0; j < 64; ++j) {
      const v4f* kr = (const v4f*)(Ks + j * HD_);
      float s = 0.f;
      #pragma unroll
      for (int i = 0; i < 16; ++i) {
        const v4f kk = kr[i];
        s += q4[i][0] * kk[0] + q4[i][1] * kk[1] +
             q4[i][2] * kk[2] + q4[i][3] * kk[3];
      }
      s = s * scale + bi;
      Ssc[j * 256 + t] = s;
      cmax = fmaxf(cmax, s);
    }
    const float nm = fmaxf(mrun, cmax);
    const float corr = __expf(mrun - nm);
    l *= corr;
    #pragma unroll
    for (int i = 0; i < 16; ++i) acc4[i] *= corr;
    for (int j = 0; j < 64; ++j) {
      const float p = __expf(Ssc[j * 256 + t] - nm);
      l += p;
      const v4f* vr = (const v4f*)(Vs + j * HD_);
      #pragma unroll
      for (int i = 0; i < 16; ++i) acc4[i] += vr[i] * p;
    }
    mrun = nm;
    __syncthreads();
  }
  const float gs = gscale_p ? tanhf(*gscale_p) : 1.f;
  const float w = gs / l;
  v4f* op = (v4f*)(Out + ((size_t)(b * T_ + t)) * D_ + h * HD_);
  if (mode == 0) {
    #pragma unroll
    for (int i = 0; i < 16; ++i) op[i] = acc4[i] * w;
  } else {
    #pragma unroll
    for (int i = 0; i < 16; ++i) op[i] += acc4[i] * w;
  }
}

// ================== reasoning-pool MHA (nh2=4, hd2=256) ====================
// grid = B*4; block = 256.
__global__ __launch_bounds__(256) void pool_attn_kernel(
    const float* __restrict__ qv, const float* __restrict__ Kp,
    const float* __restrict__ Vp, float* __restrict__ pooled)
{
  __shared__ float qs[256];
  __shared__ float probs[512];
  __shared__ float red[256];
  const int b = blockIdx.x >> 2, h2 = blockIdx.x & 3;
  const int tid = threadIdx.x;
  qs[tid] = qv[h2 * 256 + tid];
  __syncthreads();
  float sc[2], smax = -1e30f;
  for (int rr = 0; rr < 2; ++rr) {
    const int j = tid + rr * 256;
    const v4f* kp = (const v4f*)(Kp + ((size_t)(b * TR_ + j)) * D_ + h2 * 256);
    float s = 0.f;
    for (int d = 0; d < 64; ++d) {
      const v4f kk = kp[d];
      s += qs[4 * d] * kk[0] + qs[4 * d + 1] * kk[1] +
           qs[4 * d + 2] * kk[2] + qs[4 * d + 3] * kk[3];
    }
    sc[rr] = s * 0.0625f;  // 1/sqrt(256)
    smax = fmaxf(smax, sc[rr]);
  }
  red[tid] = smax; __syncthreads();
  for (int st = 128; st > 0; st >>= 1) {
    if (tid < st) red[tid] = fmaxf(red[tid], red[tid + st]);
    __syncthreads();
  }
  const float gmax = red[0]; __syncthreads();
  float ls = 0.f;
  for (int rr = 0; rr < 2; ++rr) {
    const float p = __expf(sc[rr] - gmax);
    probs[tid + rr * 256] = p; ls += p;
  }
  red[tid] = ls; __syncthreads();
  for (int st = 128; st > 0; st >>= 1) {
    if (tid < st) red[tid] += red[tid + st];
    __syncthreads();
  }
  const float tot = red[0]; __syncthreads();
  float o = 0.f;
  for (int j = 0; j < TR_; ++j)
    o += probs[j] * Vp[((size_t)(b * TR_ + j)) * D_ + h2 * 256 + tid];
  pooled[(size_t)b * D_ + h2 * 256 + tid] = o / tot;
}

// ================== small elementwise kernels ==============================
__global__ void silu_kernel(const float* __restrict__ in,
                            float* __restrict__ out, int n) {
  const int i = blockIdx.x * 256 + threadIdx.x;
  if (i < n) { const float v = in[i]; out[i] = v / (1.f + expf(-v)); }
}

__global__ void bevpool_kernel(const float* __restrict__ bev,
                               float* __restrict__ outp) {
  const int b = blockIdx.x;
  for (int c = threadIdx.x; c < D_; c += 256) {
    float s = 0.f;
    for (int t = 0; t < TBEV_; ++t) s += bev[((size_t)(b * TBEV_ + t)) * D_ + c];
    outp[(size_t)b * D_ + c] = s * (1.f / TBEV_);
  }
}

__global__ void combine_kernel(
    const float* __restrict__ x, const float* __restrict__ attn_out,
    const float* __restrict__ bres, const float* __restrict__ rres,
    const float* __restrict__ modv, const float* __restrict__ rmodv,
    const int* __restrict__ Ttraj, float* __restrict__ xnew)
{
  const int idx = blockIdx.x * 256 + threadIdx.x;
  const int r = idx >> 10, c = idx & 1023;
  const int b = r >> 8, t = r & 255;
  const float* mm = (t < *Ttraj) ? modv : rmodv;
  const float gmod = mm[(size_t)b * (6 * D_) + 2 * D_ + c]; // gate_attn slice
  const float tot = attn_out[idx] + bres[(size_t)b * D_ + c] +
                    rres[(size_t)b * D_ + c];
  xnew[idx] = x[idx] + gmod * tot;
}

// ===========================================================================
extern "C" void kernel_launch(void* const* d_in, const int* in_sizes, int n_in,
                              void* d_out, int out_size, void* d_ws,
                              size_t ws_size, hipStream_t stream) {
  (void)in_sizes; (void)n_in; (void)out_size; (void)ws_size;
  const float* x     = (const float*)d_in[0];
  const float* bev   = (const float*)d_in[1];
  const float* rsn   = (const float*)d_in[2];
  const float* cond  = (const float*)d_in[3];
  const float* rcond = (const float*)d_in[4];
  const float* P[66];
  for (int i = 0; i < 66; ++i) P[i] = (const float*)d_in[5 + i];
  const int* Tt = (const int*)d_in[71];

  enum { pre_w=0, pre_b, qn, kn, Wq, bq, Wab, bab, War, bar, Wao, bao,
         Wrab, brab, Wrar, brar, Wrao, brao, Wks, bks, Wvs, bvs, Wkb, bkb,
         Wvb, bvb, Wkr, bkr, Wvr, bvr, Wo, bo, g_p, ts, tb, tr, bs, bbp, brp_,
         rtb, rtr, rbb, rbr, Wbr1, bbr1, Wbr2, bbr2, bev_gate, rq,
         mha_in_w, mha_in_b, mha_out_w, mha_out_b, Wrp, brp, reason_gate,
         adaLN_W, adaLN_b, r_adaLN_W, r_adaLN_b, ffn_ln_w, ffn_ln_b,
         Wf1, bf1, Wf2, bf2 };

  // ---------------- workspace layout (floats), with reuse ------------------
  float* ws = (float*)d_ws;
  size_t off = 0;
  auto alloc = [&](size_t n) { size_t r = off; off += (n + 63) & ~(size_t)63; return r; };
  const size_t BT = (size_t)B_ * T_;              // 4096
  const size_t o_xnorm = alloc(BT * D_);          // -> later attn_out
  const size_t o_qbf   = alloc(BT * D_);          // -> later x_new
  const size_t o_habev = alloc(BT * 256);
  const size_t o_harsn = alloc(BT * 256);
  const size_t o_qbev  = alloc(BT * D_);          // -> later h_ln
  const size_t o_qrsn  = alloc(BT * D_);          // h1 overlay start
  const size_t o_qbase = alloc(BT * D_);
  const size_t o_ks    = alloc(BT * D_);
  const size_t o_vs    = alloc(BT * D_);
  const size_t o_kb    = alloc((size_t)B_ * TBEV_ * D_);
  const size_t o_vb    = alloc((size_t)B_ * TBEV_ * D_);
  const size_t o_kr    = alloc((size_t)B_ * TR_ * D_);   // -> later kpool
  const size_t o_vr    = alloc((size_t)B_ * TR_ * D_);   // -> later vpool
  const size_t o_accum = alloc(BT * D_);
  const size_t o_mod   = alloc((size_t)B_ * 6 * D_);
  const size_t o_rmod  = alloc((size_t)B_ * 6 * D_);
  const size_t o_siluc = alloc((size_t)B_ * D_);
  const size_t o_silur = alloc((size_t)B_ * D_);
  const size_t o_bpool = alloc((size_t)B_ * D_);
  const size_t o_bh    = alloc((size_t)B_ * 512);
  const size_t o_bres  = alloc((size_t)B_ * D_);
  const size_t o_qpool = alloc(D_);
  const size_t o_pooled= alloc((size_t)B_ * D_);
  const size_t o_pool2 = alloc((size_t)B_ * D_);
  const size_t o_rres  = alloc((size_t)B_ * D_);
  const size_t o_attno = o_xnorm;   // attn_out reuses x_norm
  const size_t o_xnew  = o_qbf;     // x_new reuses q_base_f
  const size_t o_hln   = o_qbev;    // ffn LN reuses q_bev heads
  const size_t o_h1    = o_qrsn;    // 16.78M floats over qrsn..kb (17.8M)

  const float* modp  = ws + o_mod;
  const float* rmodp = ws + o_rmod;

  auto gemm = [&](const float* A, int M, int N, int K,
                  const float* W0, const float* b0,
                  const float* W1, const float* b1,
                  float* outp, const float* src, int act,
                  const float* gate, const float* res, int slice, int headL) {
    dim3 grid((N + 127) / 128, (M + 63) / 64);
    gemm_wmma<<<grid, 256, 0, stream>>>(A, M, N, K, W0, W1, b0, b1, Tt, outp,
                                        src, act, gate, res, modp, rmodp,
                                        slice, headL);
  };

  const size_t attn_shmem = (2 * 64 * 64 + 256 * 64) * sizeof(float);

  // 1) adaLN modulation: mod = silu(cond) @ adaLN_W^T + b  (B x 6D)
  silu_kernel<<<(B_ * D_ + 255) / 256, 256, 0, stream>>>(cond, ws + o_siluc, B_ * D_);
  silu_kernel<<<(B_ * D_ + 255) / 256, 256, 0, stream>>>(rcond, ws + o_silur, B_ * D_);
  gemm(ws + o_siluc, B_, 6 * D_, D_, P[adaLN_W], P[adaLN_b], nullptr, nullptr,
       ws + o_mod, nullptr, 0, nullptr, nullptr, 0, 0);
  gemm(ws + o_silur, B_, 6 * D_, D_, P[r_adaLN_W], P[r_adaLN_b], nullptr, nullptr,
       ws + o_rmod, nullptr, 0, nullptr, nullptr, 0, 0);

  // 2) x_norm = LN(x)*(1+seg(scale_pre)) + seg(shift_pre)
  ln_mod_kernel<<<BT, 256, 0, stream>>>(x, P[pre_w], P[pre_b], modp, rmodp,
                                        1, 0, Tt, ws + o_xnorm);

  // 3) projections (head-layout stores fused into GEMM epilogue)
  gemm(ws + o_xnorm, BT, D_, D_, P[Wq], P[bq], nullptr, nullptr,
       ws + o_qbf, nullptr, 0, nullptr, nullptr, 0, 0);
  gemm(ws + o_xnorm, BT, 256, D_, P[Wab], P[bab], P[Wrab], P[brab],
       ws + o_habev, nullptr, 1, nullptr, nullptr, 0, 0);
  gemm(ws + o_xnorm, BT, 256, D_, P[War], P[bar], P[Wrar], P[brar],
       ws + o_harsn, nullptr, 1, nullptr, nullptr, 0, 0);
  gemm(ws + o_habev, BT, D_, 256, P[Wao], P[bao], P[Wrao], P[brao],
       ws + o_qbev, ws + o_qbf, 0, nullptr, nullptr, 0, T_);
  gemm(ws + o_harsn, BT, D_, 256, P[Wao], P[bao], P[Wrao], P[brao],
       ws + o_qrsn, ws + o_qbf, 0, nullptr, nullptr, 0, T_);
  gemm(ws + o_xnorm, BT, D_, D_, P[Wks], P[bks], nullptr, nullptr,
       ws + o_ks, nullptr, 0, nullptr, nullptr, 0, T_);
  gemm(ws + o_xnorm, BT, D_, D_, P[Wvs], P[bvs], nullptr, nullptr,
       ws + o_vs, nullptr, 0, nullptr, nullptr, 0, T_);
  gemm(bev, B_ * TBEV_, D_, D_, P[Wkb], P[bkb], nullptr, nullptr,
       ws + o_kb, nullptr, 0, nullptr, nullptr, 0, TBEV_);
  gemm(bev, B_ * TBEV_, D_, D_, P[Wvb], P[bvb], nullptr, nullptr,
       ws + o_vb, nullptr, 0, nullptr, nullptr, 0, TBEV_);
  gemm(rsn, B_ * TR_, D_, D_, P[Wkr], P[bkr], nullptr, nullptr,
       ws + o_kr, nullptr, 0, nullptr, nullptr, 0, TR_);
  gemm(rsn, B_ * TR_, D_, D_, P[Wvr], P[bvr], nullptr, nullptr,
       ws + o_vr, nullptr, 0, nullptr, nullptr, 0, TR_);

  // 4) RMS + RoPE
  rms_rope_kernel<<<B_ * H_ * T_, 64, 0, stream>>>(ws + o_qbf, ws + o_qbase,
                                                   T_, P[qn], 1, 1);
  rms_rope_kernel<<<B_ * H_ * T_, 64, 0, stream>>>(ws + o_qbev, ws + o_qbev,
                                                   T_, P[qn], 0, 0);
  rms_rope_kernel<<<B_ * H_ * T_, 64, 0, stream>>>(ws + o_qrsn, ws + o_qrsn,
                                                   T_, P[qn], 0, 0);
  rms_rope_kernel<<<B_ * H_ * T_, 64, 0, stream>>>(ws + o_ks, ws + o_ks,
                                                   T_, P[kn], 1, 0);
  rms_rope_kernel<<<B_ * H_ * TBEV_, 64, 0, stream>>>(ws + o_kb, ws + o_kb,
                                                      TBEV_, P[kn], 1, 0);
  rms_rope_kernel<<<B_ * H_ * TR_, 64, 0, stream>>>(ws + o_kr, ws + o_kr,
                                                    TR_, P[kn], 1, 0);

  // 5) three attentions accumulated into o_accum
  attn_kernel<<<B_ * H_, 256, attn_shmem, stream>>>(
      ws + o_qbase, ws + o_ks, ws + o_vs, T_, P[ts], P[bs], nullptr, nullptr,
      Tt, nullptr, 0, ws + o_accum);
  attn_kernel<<<B_ * H_, 256, attn_shmem, stream>>>(
      ws + o_qbev, ws + o_kb, ws + o_vb, TBEV_, P[tb], P[bbp], P[rtb], P[rbb],
      Tt, nullptr, 1, ws + o_accum);
  attn_kernel<<<B_ * H_, 256, attn_shmem, stream>>>(
      ws + o_qrsn, ws + o_kr, ws + o_vr, TR_, P[tr], P[brp_], P[rtr], P[rbr],
      Tt, P[g_p], 1, ws + o_accum);

  // 6) attention output projection
  gemm(ws + o_accum, BT, D_, D_, P[Wo], P[bo], nullptr, nullptr,
       ws + o_attno, nullptr, 0, nullptr, nullptr, 0, 0);

  // 7) BEV-pool residual
  bevpool_kernel<<<B_, 256, 0, stream>>>(bev, ws + o_bpool);
  gemm(ws + o_bpool, B_, 512, D_, P[Wbr1], P[bbr1], nullptr, nullptr,
       ws + o_bh, nullptr, 2, nullptr, nullptr, 0, 0);
  gemm(ws + o_bh, B_, D_, 512, P[Wbr2], P[bbr2], nullptr, nullptr,
       ws + o_bres, nullptr, 0, P[bev_gate], nullptr, 0, 0);

  // 8) reasoning-pool residual (kpool/vpool overlay dead kr/vr)
  gemm(P[rq], 1, D_, D_, P[mha_in_w], P[mha_in_b], nullptr, nullptr,
       ws + o_qpool, nullptr, 0, nullptr, nullptr, 0, 0);
  gemm(rsn, B_ * TR_, D_, D_, P[mha_in_w] + (size_t)D_ * D_, P[mha_in_b] + D_,
       nullptr, nullptr, ws + o_kr, nullptr, 0, nullptr, nullptr, 0, 0);
  gemm(rsn, B_ * TR_, D_, D_, P[mha_in_w] + 2 * (size_t)D_ * D_,
       P[mha_in_b] + 2 * D_, nullptr, nullptr, ws + o_vr, nullptr, 0, nullptr,
       nullptr, 0, 0);
  pool_attn_kernel<<<B_ * 4, 256, 0, stream>>>(ws + o_qpool, ws + o_kr,
                                               ws + o_vr, ws + o_pooled);
  gemm(ws + o_pooled, B_, D_, D_, P[mha_out_w], P[mha_out_b], nullptr, nullptr,
       ws + o_pool2, nullptr, 0, nullptr, nullptr, 0, 0);
  gemm(ws + o_pool2, B_, D_, D_, P[Wrp], P[brp], nullptr, nullptr,
       ws + o_rres, nullptr, 0, P[reason_gate], nullptr, 0, 0);

  // 9) x_new = x + seg(gate_attn)*(attn_out + bev_res + rsn_res)
  combine_kernel<<<(BT * D_) / 256, 256, 0, stream>>>(
      x, ws + o_attno, ws + o_bres, ws + o_rres, modp, rmodp, Tt, ws + o_xnew);

  // 10) FFN: LN -> gelu(Wf1) -> Wf2, gated residual straight into d_out
  ln_mod_kernel<<<BT, 256, 0, stream>>>(ws + o_xnew, P[ffn_ln_w], P[ffn_ln_b],
                                        modp, rmodp, 4, 3, Tt, ws + o_hln);
  gemm(ws + o_hln, BT, FF_, D_, P[Wf1], P[bf1], nullptr, nullptr,
       ws + o_h1, nullptr, 2, nullptr, nullptr, 0, 0);
  gemm(ws + o_h1, BT, D_, FF_, P[Wf2], P[bf2], nullptr, nullptr,
       (float*)d_out, nullptr, 0, nullptr, ws + o_xnew, 5, 0);
}
